// PointerLayer_8177617731971
// MI455X (gfx1250) — matrix operand order repaired and checked by reading, not compiled, not run on hardware
//
#include <hip/hip_runtime.h>
#include <hip/hip_bf16.h>
#include <math.h>

// ---------------------------------------------------------------------------
// Problem constants (from reference): B=64, S=256, H=512, W=512
// ---------------------------------------------------------------------------
#define S_LEN 256
#define BATCH 64
#define HID   512
#define WID   512
#define G4H   2048   // 4*H

typedef __attribute__((ext_vector_type(16))) __bf16 v16bf;
typedef __attribute__((ext_vector_type(8)))  __bf16 v8bf;
typedef __attribute__((ext_vector_type(8)))  float  v8f;

__device__ __forceinline__ unsigned short f32_bf16(float x) {
  union { float f; unsigned u; } v; v.f = x;
  unsigned r = v.u + 0x7FFFu + ((v.u >> 16) & 1u);   // round-to-nearest-even
  return (unsigned short)(r >> 16);
}
__device__ __forceinline__ float bf16_f32(unsigned short h) {
  union { unsigned u; float f; } v; v.u = ((unsigned)h) << 16;
  return v.f;
}
__device__ __forceinline__ float sigmoidf_(float x) {
  return 1.0f / (1.0f + __expf(-x));
}

// ---------------------------------------------------------------------------
// Generic WMMA GEMM: C[M,N] = A[M,K] @ W[N,K]^T (+ bias[N])
// A, W bf16 row-major (lda/ldb = K). C f32 row-major (ldc = N).
// Block = 256 threads = 8 waves (wave32), 2(M) x 4(N); wave tile 32x32 =
// 2x2 v_wmma_f32_16x16x32_bf16. Block tile 64x128. Grid exactly covers M,N.
// ---------------------------------------------------------------------------
__global__ __launch_bounds__(256) void wmma_gemm(
    const __bf16* __restrict__ A, int lda,
    const __bf16* __restrict__ W, int ldb,
    const float* __restrict__ bias,
    float* __restrict__ C, int ldc,
    int K)
{
  const int lane = threadIdx.x & 31;
  const int wave = threadIdx.x >> 5;
  const int m0 = blockIdx.y * 64  + (wave >> 2) * 32;
  const int n0 = blockIdx.x * 128 + (wave &  3) * 32;

  const int cn = lane & 15;                 // C: N = cn(+16*half), M = r + cr
  const int cr = (lane < 16) ? 0 : 8;

  v8f acc[2][2];
#pragma unroll
  for (int mi = 0; mi < 2; ++mi)
#pragma unroll
    for (int ni = 0; ni < 2; ++ni)
#pragma unroll
      for (int r = 0; r < 8; ++r)
        acc[mi][ni][r] = 0.0f;

  const int arow = lane & 15;
  const int ak   = (lane < 16) ? 0 : 16;    // A lane K chunk of 16
  const int bk   = (lane < 16) ? 0 : 16;    // B lane K chunk of 16

  for (int k0 = 0; k0 < K; k0 += 32) {
    v16bf a[2], b[2];
#pragma unroll
    for (int mi = 0; mi < 2; ++mi) {
      const __bf16* ap = A + (size_t)(m0 + mi * 16 + arow) * lda + k0 + ak;
      v8bf lo = *(const v8bf*)ap;
      v8bf hi = *(const v8bf*)(ap + 8);
      a[mi] = __builtin_shufflevector(lo, hi, 0,1,2,3,4,5,6,7,8,9,10,11,12,13,14,15);
    }
#pragma unroll
    for (int ni = 0; ni < 2; ++ni)
      b[ni] = *(const v16bf*)(W + (size_t)(n0 + ni * 16 + cn) * ldb + k0 + bk);
#pragma unroll
    for (int mi = 0; mi < 2; ++mi)
#pragma unroll
      for (int ni = 0; ni < 2; ++ni)
        acc[mi][ni] = __builtin_amdgcn_wmma_f32_16x16x32_bf16(
            false, a[mi], false, b[ni], (short)0, acc[mi][ni], false, false);
  }

#pragma unroll
  for (int mi = 0; mi < 2; ++mi)
#pragma unroll
    for (int ni = 0; ni < 2; ++ni) {
      const int n = n0 + ni * 16 + cn;
      const float bv = bias ? bias[n] : 0.0f;
#pragma unroll
      for (int r = 0; r < 8; ++r)
        C[(size_t)(m0 + mi * 16 + cr + r) * ldc + n] = acc[mi][ni][r] + bv;
    }
}

// ---------------------------------------------------------------------------
// Fused recurrent step: g = xe[t] + h_in @ Whh_p^T, then LSTM cell, all in one
// kernel.  Weights are gate-permuted (row n' = 4*h + gate, gate order i,f,g,o)
// so one 64x128 block tile holds all 4 gates of 32 hidden units for the full
// batch.  Accumulator tile staged in LDS, cell applied in epilogue.
// Grid: (G4H/128) blocks, 256 threads.  M = BATCH = 64 (exactly one tile).
// ---------------------------------------------------------------------------
__global__ __launch_bounds__(256) void wmma_gemm_lstm(
    const __bf16* __restrict__ A,          // h_in  [B,H] bf16
    const __bf16* __restrict__ Wp,         // permuted Whh [4H,H] bf16
    const float* __restrict__ Cin,         // xe_p + t*B*4H, [B,4H] f32 (perm cols)
    float* __restrict__ c,                 // [B*H] f32 (enc: rw, dec: read-only)
    unsigned short* __restrict__ hout,     // [B*H] bf16
    unsigned short* __restrict__ enc_row,  // [B*H] bf16 or nullptr
    int is_dec)
{
  __shared__ float sm[64 * 128];           // 32 KB gate tile
  const int lane = threadIdx.x & 31;
  const int wave = threadIdx.x >> 5;
  const int m0  = (wave >> 2) * 32;        // 0 or 32 (M=64 total)
  const int n0w = (wave & 3) * 32;         // N offset within block tile
  const int n0  = blockIdx.x * 128 + n0w;

  const int cn = lane & 15;
  const int cr = (lane < 16) ? 0 : 8;

  v8f acc[2][2];
#pragma unroll
  for (int mi = 0; mi < 2; ++mi)
#pragma unroll
    for (int ni = 0; ni < 2; ++ni)
#pragma unroll
      for (int r = 0; r < 8; ++r)
        acc[mi][ni][r] = Cin[(size_t)(m0 + mi * 16 + cr + r) * G4H + n0 + ni * 16 + cn];

  const int arow = lane & 15;
  const int ak   = (lane < 16) ? 0 : 16;
  const int bk   = (lane < 16) ? 0 : 16;

  for (int k0 = 0; k0 < HID; k0 += 32) {
    v16bf a[2], b[2];
#pragma unroll
    for (int mi = 0; mi < 2; ++mi) {
      const __bf16* ap = A + (size_t)(m0 + mi * 16 + arow) * HID + k0 + ak;
      v8bf lo = *(const v8bf*)ap;
      v8bf hi = *(const v8bf*)(ap + 8);
      a[mi] = __builtin_shufflevector(lo, hi, 0,1,2,3,4,5,6,7,8,9,10,11,12,13,14,15);
    }
#pragma unroll
    for (int ni = 0; ni < 2; ++ni)
      b[ni] = *(const v16bf*)(Wp + (size_t)(n0 + ni * 16 + cn) * HID + k0 + bk);
#pragma unroll
    for (int mi = 0; mi < 2; ++mi)
#pragma unroll
      for (int ni = 0; ni < 2; ++ni)
        acc[mi][ni] = __builtin_amdgcn_wmma_f32_16x16x32_bf16(
            false, a[mi], false, b[ni], (short)0, acc[mi][ni], false, false);
  }

  // stage gate tile to LDS
#pragma unroll
  for (int mi = 0; mi < 2; ++mi)
#pragma unroll
    for (int ni = 0; ni < 2; ++ni)
#pragma unroll
      for (int r = 0; r < 8; ++r)
        sm[(m0 + mi * 16 + cr + r) * 128 + n0w + ni * 16 + cn] = acc[mi][ni][r];
  __syncthreads();

  // epilogue: LSTM cell on 64 x 32 (b, h) pairs; gates adjacent (i,f,g,o)
  const int h0 = blockIdx.x * 32;
#pragma unroll
  for (int j = 0; j < 8; ++j) {
    const int p  = threadIdx.x + j * 256;     // 0..2047
    const int b  = p >> 5;
    const int hh = p & 31;
    const float4 gv = *(const float4*)&sm[b * 128 + hh * 4];
    const float ig = sigmoidf_(gv.x);
    const float fg = sigmoidf_(gv.y);
    const float gg = tanhf(gv.z);
    const float og = sigmoidf_(gv.w);
    const int ci = b * HID + h0 + hh;
    const float cnew = fg * c[ci] + ig * gg;
    if (!is_dec) c[ci] = cnew;               // decoder cell stays at cT (reference)
    const unsigned short hb = f32_bf16(og * tanhf(cnew));
    hout[ci] = hb;
    if (enc_row) enc_row[ci] = hb;
  }
}

// ---------------------------------------------------------------------------
// Elementwise helpers
// ---------------------------------------------------------------------------
__global__ void cvt_bf16(const float* __restrict__ in, unsigned short* __restrict__ out, int n) {
  int i = blockIdx.x * blockDim.x + threadIdx.x;
  if (i < n) out[i] = f32_bf16(in[i]);
}

// gate-permute + convert weight rows: out row n' = 4*h + gate <- in row gate*H + h
__global__ void cvt_bf16_perm(const float* __restrict__ in, unsigned short* __restrict__ out,
                              int K) {
  int i = blockIdx.x * blockDim.x + threadIdx.x;   // G4H*K threads
  int k = i % K;
  int n = i / K;
  int gate = n & 3, h = n >> 2;
  out[i] = f32_bf16(in[(size_t)(gate * HID + h) * K + k]);
}

__global__ void bias_perm(const float* __restrict__ a, const float* __restrict__ b,
                          float* __restrict__ o) {
  int n = blockIdx.x * blockDim.x + threadIdx.x;   // G4H threads
  int gate = n & 3, h = n >> 2;
  int on = gate * HID + h;
  o[n] = a[on] + b[on];
}

// x[B,S,S] -> xtm[S,B,256] (time-major) bf16
__global__ void xpose_cvt(const float* __restrict__ x, unsigned short* __restrict__ xtm) {
  int i = blockIdx.x * blockDim.x + threadIdx.x;   // S*B*256 threads
  int k = i & 255, rem = i >> 8;
  int b = rem & 63, t = rem >> 6;
  xtm[i] = f32_bf16(x[((size_t)b * S_LEN + t) * S_LEN + k]);
}

__global__ void init_state(float* __restrict__ c, unsigned short* __restrict__ h) {
  int i = blockIdx.x * blockDim.x + threadIdx.x;   // B*H threads
  c[i] = 0.0f;
  h[i] = 0;
}

// ---------------------------------------------------------------------------
// Fused decoder attention for step t (one block per batch row, 8 waves):
//   hw2[w]   = sum_k h_new[b,k] * w2[w,k]          (w2 kept f32)
//   score[s] = sum_w vt[w] * tanh(e1[s,b,w] + hw2[w])
//   out[b,t,:] = log_softmax(score)
// e1 (33.5 MB) is resident in the 192 MB L2 across all 256 steps.
// ---------------------------------------------------------------------------
__global__ __launch_bounds__(256) void attn_fused(
    const float* __restrict__ e1,          // [S,B,W]
    const unsigned short* __restrict__ hbf,// [B,H] bf16 (h_new)
    const float* __restrict__ w2,          // [W,H] f32
    const float* __restrict__ vt,          // [W]
    float* __restrict__ out,               // [B,S,S]
    int t)
{
  __shared__ float hs[HID];
  __shared__ float hw2s[WID];
  __shared__ float sc[S_LEN];
  __shared__ float red0[8];
  __shared__ float red1[8];
  const int b = blockIdx.x;
  const int tid = threadIdx.x;
  const int lane = tid & 31;
  const int wave = tid >> 5;

  for (int k = tid; k < HID; k += 256)
    hs[k] = bf16_f32(hbf[(size_t)b * HID + k]);
  __syncthreads();

  for (int w = wave; w < WID; w += 8) {
    const float* wr = w2 + (size_t)w * HID;
    float s = 0.0f;
    for (int k = lane; k < HID; k += 32)
      s += hs[k] * wr[k];
#pragma unroll
    for (int off = 16; off; off >>= 1)
      s += __shfl_xor(s, off, 32);
    if (lane == 0) hw2s[w] = s;
  }
  __syncthreads();

  for (int s = wave; s < S_LEN; s += 8) {
    const float* erow = e1 + ((size_t)s * BATCH + b) * WID;
    float sum = 0.0f;
    for (int w = lane; w < WID; w += 32)
      sum += vt[w] * tanhf(erow[w] + hw2s[w]);
#pragma unroll
    for (int off = 16; off; off >>= 1)
      sum += __shfl_xor(sum, off, 32);
    if (lane == 0) sc[s] = sum;
  }
  __syncthreads();

  const float v = sc[tid];
  float m = v;
#pragma unroll
  for (int off = 16; off; off >>= 1)
    m = fmaxf(m, __shfl_xor(m, off, 32));
  if (lane == 0) red0[wave] = m;
  __syncthreads();
  float gmax = red0[0];
#pragma unroll
  for (int i = 1; i < 8; ++i) gmax = fmaxf(gmax, red0[i]);

  float ssum = __expf(v - gmax);
#pragma unroll
  for (int off = 16; off; off >>= 1)
    ssum += __shfl_xor(ssum, off, 32);
  if (lane == 0) red1[wave] = ssum;
  __syncthreads();
  float tot = 0.0f;
#pragma unroll
  for (int i = 0; i < 8; ++i) tot += red1[i];

  out[(size_t)b * S_LEN * S_LEN + (size_t)t * S_LEN + tid] = v - gmax - __logf(tot);
}

// ---------------------------------------------------------------------------
// Host launcher
// ---------------------------------------------------------------------------
extern "C" void kernel_launch(void* const* d_in, const int* in_sizes, int n_in,
                              void* d_out, int out_size, void* d_ws, size_t ws_size,
                              hipStream_t stream) {
  (void)in_sizes; (void)n_in; (void)out_size; (void)ws_size;
  const float* x      = (const float*)d_in[0];
  const float* encWih = (const float*)d_in[1];
  const float* encWhh = (const float*)d_in[2];
  const float* encbih = (const float*)d_in[3];
  const float* encbhh = (const float*)d_in[4];
  const float* decWih = (const float*)d_in[5];
  const float* decWhh = (const float*)d_in[6];
  const float* decbih = (const float*)d_in[7];
  const float* decbhh = (const float*)d_in[8];
  const float* w1     = (const float*)d_in[9];
  const float* w2     = (const float*)d_in[10];
  const float* vt     = (const float*)d_in[11];
  float* out = (float*)d_out;

  char* base = (char*)d_ws;
  size_t off = 0;
  auto alloc = [&](size_t bytes) -> void* {
    void* p = base + off;
    off += (bytes + 255) & ~(size_t)255;
    return p;
  };

  unsigned short* xtm     = (unsigned short*)alloc((size_t)S_LEN * BATCH * 256 * 2);
  unsigned short* eWih_p  = (unsigned short*)alloc((size_t)G4H * 256 * 2);
  unsigned short* eWhh_p  = (unsigned short*)alloc((size_t)G4H * HID * 2);
  unsigned short* dWih_p  = (unsigned short*)alloc((size_t)G4H * 256 * 2);
  unsigned short* dWhh_p  = (unsigned short*)alloc((size_t)G4H * HID * 2);
  unsigned short* w1_bf   = (unsigned short*)alloc((size_t)WID * HID * 2);
  float* ebias_p          = (float*)alloc((size_t)G4H * 4);
  float* dbias_p          = (float*)alloc((size_t)G4H * 4);
  float* xe_enc           = (float*)alloc((size_t)S_LEN * BATCH * G4H * 4);  // [S,B,4H] perm
  float* xe_dec           = (float*)alloc((size_t)S_LEN * BATCH * G4H * 4);
  unsigned short* encout  = (unsigned short*)alloc((size_t)S_LEN * BATCH * HID * 2);
  float* e1               = (float*)alloc((size_t)S_LEN * BATCH * WID * 4);
  float* c_ws             = (float*)alloc((size_t)BATCH * HID * 4);
  unsigned short* hbuf0   = (unsigned short*)alloc((size_t)BATCH * HID * 2);
  unsigned short* hbuf1   = (unsigned short*)alloc((size_t)BATCH * HID * 2);
  unsigned short* hb[2] = { hbuf0, hbuf1 };

  // --- stage 0: convert (gate-permuted) weights / biases / inputs ----------
  cvt_bf16_perm<<<(G4H * 256) / 256, 256, 0, stream>>>(encWih, eWih_p, 256);
  cvt_bf16_perm<<<(G4H * HID) / 256, 256, 0, stream>>>(encWhh, eWhh_p, HID);
  cvt_bf16_perm<<<(G4H * 256) / 256, 256, 0, stream>>>(decWih, dWih_p, 256);
  cvt_bf16_perm<<<(G4H * HID) / 256, 256, 0, stream>>>(decWhh, dWhh_p, HID);
  cvt_bf16<<<(WID * HID) / 256, 256, 0, stream>>>(w1, w1_bf, WID * HID);
  bias_perm<<<G4H / 256, 256, 0, stream>>>(encbih, encbhh, ebias_p);
  bias_perm<<<G4H / 256, 256, 0, stream>>>(decbih, decbhh, dbias_p);
  xpose_cvt<<<(S_LEN * BATCH * 256) / 256, 256, 0, stream>>>(x, xtm);

  // --- stage 1: hoisted input projections (big parallel WMMA GEMMs) --------
  // xe = Xtm[16384,256] @ Wih_p[2048,256]^T + bias_p   (gate-permuted columns)
  {
    dim3 grid(G4H / 128, (S_LEN * BATCH) / 64);
    wmma_gemm<<<grid, 256, 0, stream>>>((const __bf16*)xtm, 256,
                                        (const __bf16*)eWih_p, 256,
                                        ebias_p, xe_enc, G4H, 256);
    wmma_gemm<<<grid, 256, 0, stream>>>((const __bf16*)xtm, 256,
                                        (const __bf16*)dWih_p, 256,
                                        dbias_p, xe_dec, G4H, 256);
  }

  init_state<<<(BATCH * HID) / 256, 256, 0, stream>>>(c_ws, hb[0]);

  // --- stage 2: encoder recurrence (fused GEMM + LSTM cell, ping-pong h) ---
  for (int t = 0; t < S_LEN; ++t) {
    wmma_gemm_lstm<<<G4H / 128, 256, 0, stream>>>(
        (const __bf16*)hb[t & 1], (const __bf16*)eWhh_p,
        xe_enc + (size_t)t * BATCH * G4H, c_ws,
        hb[(t + 1) & 1], encout + (size_t)t * BATCH * HID, 0);
  }
  // after loop: hT in hb[0] (256 steps), cT in c_ws

  // --- stage 3: e1 = enc_out[16384,512] @ w1[512,512]^T ---------------------
  {
    dim3 grid(WID / 128, (S_LEN * BATCH) / 64);
    wmma_gemm<<<grid, 256, 0, stream>>>((const __bf16*)encout, HID,
                                        (const __bf16*)w1_bf, HID,
                                        nullptr, e1, WID, HID);
  }

  // --- stage 4: decoder recurrence + fused attention ------------------------
  for (int t = 0; t < S_LEN; ++t) {
    wmma_gemm_lstm<<<G4H / 128, 256, 0, stream>>>(
        (const __bf16*)hb[t & 1], (const __bf16*)dWhh_p,
        xe_dec + (size_t)t * BATCH * G4H, c_ws,
        hb[(t + 1) & 1], nullptr, 1);
    attn_fused<<<BATCH, 256, 0, stream>>>(e1, hb[(t + 1) & 1], w2, vt, out, t);
  }
}
